// Img_Point_Fusion_Net_44367012168418
// MI455X (gfx1250) — matrix operand order repaired and checked by reading, not compiled, not run on hardware
//
#include <hip/hip_runtime.h>

typedef __attribute__((ext_vector_type(16))) _Float16 v16h;
typedef __attribute__((ext_vector_type(8)))  _Float16 v8h;
typedef __attribute__((ext_vector_type(8)))  float    v8f;

// ---------------------------------------------------------------------------
// GEMM via WMMA f16->f32:  Y[b][m][o] = sum_k W[b][o][k] * X[b][m][k]
//   X : f16, layout [b][M][Kpad], Kpad % 32 == 0, zero-padded beyond real K
//   W : f32, row-major [O][Kw] at W + b*wSb (wSb==0 => shared weights)
//   Y : f32, layout [b][M][O]
// Block = 128 threads (4 waves). Each wave computes a 16(O) x 32(M) tile
// (one A fragment reused for two B tiles -> 2 WMMA per K-step). The 32(M) x
// 32(K) X tile is staged in double-buffered LDS: one barrier per K-step, the
// next tile's global_load_b128 is issued before the barrier (sw pipeline).
// GUARD=true adds per-element k<Kw masking on W (only needed when Kw%32 != 0).
// ---------------------------------------------------------------------------
template <bool GUARD>
__global__ void __launch_bounds__(128)
k_gemm(const float* __restrict__ W, long long wSb, int Kw,
       const _Float16* __restrict__ X, int Kpad,
       float* __restrict__ Y, int O, int M)
{
    __shared__ __align__(16) _Float16 lds[2][32 * 40];  // 32 rows(m) x 32 k, stride 40
    const int b    = blockIdx.z;
    const int mt   = blockIdx.x * 32;
    const int wave = threadIdx.x >> 5;
    const int lane = threadIdx.x & 31;
    const int ot   = blockIdx.y * 64 + wave * 16;
    const bool act = (ot < O);           // wave-uniform
    const bool t1  = (mt + 16 < M);      // second M tile valid, wave-uniform
    const int hi   = lane >> 4;
    const int lo   = lane & 15;

    const float*    Wb = W + (long long)b * wSb;
    const _Float16* Xb = X + (long long)b * M * Kpad;

    // staging: 128 threads x 16B = 32 rows x 32 k halves
    const int sm   = threadIdx.x >> 2;                      // 0..31
    const int skq  = threadIdx.x & 3;                       // 0..3
    const int srw  = (mt + sm < M) ? (mt + sm) : (M - 1);   // clamp (M%32 tail)
    const _Float16* srow = Xb + (long long)srw * Kpad + skq * 8;
    const int sdst = sm * 40 + skq * 8;

    v8f acc0 = {}, acc1 = {};

    // prologue: stage first K tile into buffer 0
    *(v8h*)&lds[0][sdst] = *(const v8h*)srow;

    for (int k0 = 0; k0 < Kpad; k0 += 32) {
        const int  cur  = (k0 >> 5) & 1;
        const bool more = (k0 + 32 < Kpad);
        v8h nxt;
        if (more) nxt = *(const v8h*)(srow + k0 + 32);   // prefetch next tile
        __syncthreads();                                  // buf[cur] ready, buf[cur^1] free
        if (more) *(v8h*)&lds[cur ^ 1][sdst] = nxt;

        if (act) {
            // ---- A = W tile 16(o) x 32(k), f32 -> f16 convert on load ----
            // lane layout (16-bit A 16x32): VGPR0-3 -> K = g*8+0..7,
            //                               VGPR4-7 -> K = 16+g*8+0..7 (g = hi)
            v16h a;
            const float* wr = Wb + (long long)(ot + lo) * Kw;
            const int kA = k0 + hi * 8;
            const int kB = k0 + 16 + hi * 8;
            if (!GUARD) {
                const float4* p = (const float4*)(wr + kA);
                float4 f0 = p[0], f1 = p[1];
                a[0] = (_Float16)f0.x; a[1] = (_Float16)f0.y;
                a[2] = (_Float16)f0.z; a[3] = (_Float16)f0.w;
                a[4] = (_Float16)f1.x; a[5] = (_Float16)f1.y;
                a[6] = (_Float16)f1.z; a[7] = (_Float16)f1.w;
                const float4* q = (const float4*)(wr + kB);
                float4 g0 = q[0], g1 = q[1];
                a[8]  = (_Float16)g0.x; a[9]  = (_Float16)g0.y;
                a[10] = (_Float16)g0.z; a[11] = (_Float16)g0.w;
                a[12] = (_Float16)g1.x; a[13] = (_Float16)g1.y;
                a[14] = (_Float16)g1.z; a[15] = (_Float16)g1.w;
            } else {
                #pragma unroll
                for (int i = 0; i < 8; ++i)
                    a[i] = (kA + i < Kw) ? (_Float16)wr[kA + i] : (_Float16)0.f;
                #pragma unroll
                for (int i = 0; i < 8; ++i)
                    a[8 + i] = (kB + i < Kw) ? (_Float16)wr[kB + i] : (_Float16)0.f;
            }

            // ---- B tiles from LDS ----
            // lane half g = hi: VGPR0-3 -> K = g*8+0..7,
            // VGPR4-7 -> K = 16+g*8+0..7 (matches A).
            {
                const v8h* ls = (const v8h*)&lds[cur][lo * 40 + hi * 8];
                v8h b0 = ls[0], b1 = ls[2];
                v16h bv;
                #pragma unroll
                for (int i = 0; i < 8; ++i) { bv[i] = b0[i]; bv[8 + i] = b1[i]; }
                acc0 = __builtin_amdgcn_wmma_f32_16x16x32_f16(
                    false, a, false, bv, (short)0, acc0, false, false);
            }
            if (t1) {
                const v8h* ls = (const v8h*)&lds[cur][(16 + lo) * 40 + hi * 8];
                v8h b0 = ls[0], b1 = ls[2];
                v16h bv;
                #pragma unroll
                for (int i = 0; i < 8; ++i) { bv[i] = b0[i]; bv[8 + i] = b1[i]; }
                acc1 = __builtin_amdgcn_wmma_f32_16x16x32_f16(
                    false, a, false, bv, (short)0, acc1, false, false);
            }
        }
    }

    if (act) {
        // D layout: VGPR r -> row o = ot + r + 8*hi, col m = tile_m + lo
        float* yp = Y + ((long long)b * M + mt + lo) * O + ot + hi * 8;
        *(float4*)yp       = make_float4(acc0[0], acc0[1], acc0[2], acc0[3]);
        *(float4*)(yp + 4) = make_float4(acc0[4], acc0[5], acc0[6], acc0[7]);
        if (t1) {
            float* yq = Y + ((long long)b * M + mt + 16 + lo) * O + ot + hi * 8;
            *(float4*)yq       = make_float4(acc1[0], acc1[1], acc1[2], acc1[3]);
            *(float4*)(yq + 4) = make_float4(acc1[4], acc1[5], acc1[6], acc1[7]);
        }
    }
}

// ---------------------------------------------------------------------------
// BatchNorm statistics: per-channel mean / rstd over rows = B*M
// Y layout [row][O], stats[o*2]=mean, stats[o*2+1]=rsqrt(var+eps)
// ---------------------------------------------------------------------------
__global__ void k_bn_stats(const float* __restrict__ Y, int rows, int O,
                           float* __restrict__ stats)
{
    int o = blockIdx.x * blockDim.x + threadIdx.x;
    if (o >= O) return;
    float s = 0.f, s2 = 0.f;
    for (int r = 0; r < rows; ++r) {
        float v = Y[(long long)r * O + o];
        s += v; s2 += v * v;
    }
    float m   = s / rows;
    float var = s2 / rows - m * m;
    stats[o * 2]     = m;
    stats[o * 2 + 1] = rsqrtf(var + 1e-5f);
}

// BN + ReLU, write next-layer f16 activation [row][Opad] (zero-fill pad)
__global__ void k_bn_relu_pack(const float* __restrict__ Y,
                               const float* __restrict__ stats,
                               int rows, int O, _Float16* __restrict__ Xo, int Opad)
{
    long long i = (long long)blockIdx.x * blockDim.x + threadIdx.x;
    long long tot = (long long)rows * Opad;
    if (i >= tot) return;
    int o = (int)(i % Opad);
    long long r = i / Opad;
    float v = 0.f;
    if (o < O) {
        float y = Y[r * O + o];
        v = fmaxf(0.f, (y - stats[o * 2]) * stats[o * 2 + 1]);
    }
    Xo[i] = (_Float16)v;
}

// BN + ReLU + transpose [b][M][O] -> [b][O][M]  (final img_out)
__global__ void k_bn_relu_tr(const float* __restrict__ Y,
                             const float* __restrict__ stats,
                             int M, int O, float* __restrict__ Out, int Bc)
{
    long long i = (long long)blockIdx.x * blockDim.x + threadIdx.x;
    long long tot = (long long)Bc * O * M;
    if (i >= tot) return;
    int m = (int)(i % M);
    int o = (int)((i / M) % O);
    int b = (int)(i / ((long long)M * O));
    float y = Y[((long long)b * M + m) * O + o];
    Out[i] = fmaxf(0.f, (y - stats[o * 2]) * stats[o * 2 + 1]);
}

// Softmax over contiguous C per row, one wave per row, f16 output [row][Cpad]
__global__ void k_softmax_pack(const float* __restrict__ Y, int C,
                               _Float16* __restrict__ Xo, int Cpad)
{
    int row = blockIdx.x;
    const float* y = Y + (long long)row * C;
    _Float16*   xo = Xo + (long long)row * Cpad;
    int l = threadIdx.x;
    float mx = -1e30f;
    for (int c = l; c < C; c += 32) mx = fmaxf(mx, y[c]);
    for (int s = 16; s > 0; s >>= 1) mx = fmaxf(mx, __shfl_xor(mx, s));
    float sum = 0.f;
    for (int c = l; c < C; c += 32) sum += __expf(y[c] - mx);
    for (int s = 16; s > 0; s >>= 1) sum += __shfl_xor(sum, s);
    float inv = 1.f / sum;
    for (int c = l; c < C; c += 32) xo[c] = (_Float16)(__expf(y[c] - mx) * inv);
    for (int c = C + l; c < Cpad; c += 32) xo[c] = (_Float16)0.f;
}

// ---------------------------------------------------------------------------
// Generic concat / transpose / broadcast / upsample packer -> f16 [b][M][Cpad]
// modes: 0 f32 [b][C][M] transpose        1 f32 [b][M][C] direct
//        2 f32 [b][C] broadcast over m    3 f16 [b][M/4][p0] 2x nearest-upsample
//        4 f32 [b][C][4M] transpose, ::2,::2 subsample (src is 2H x 2W)
// ---------------------------------------------------------------------------
struct Src { const void* p; int C; int mode; int p0; };

__device__ __forceinline__ float pk_fetch(const Src s, int b, int m, int c,
                                          int M, int W)
{
    switch (s.mode) {
    case 0: return ((const float*)s.p)[((long long)b * s.C + c) * M + m];
    case 1: return ((const float*)s.p)[((long long)b * M + m) * s.C + c];
    case 2: return ((const float*)s.p)[(long long)b * s.C + c];
    case 3: {
        int h = m / W, w = m % W;
        int mp = (h >> 1) * (W >> 1) + (w >> 1);
        return (float)((const _Float16*)s.p)[((long long)b * (M >> 2) + mp) * s.p0 + c];
    }
    default: {
        int h = m / W, w = m % W;
        long long ms = (long long)(2 * h) * (2 * W) + 2 * w;
        return ((const float*)s.p)[((long long)b * s.C + c) * ((long long)M * 4) + ms];
    }
    }
}

__global__ void k_pack(Src s0, Src s1, Src s2, Src s3,
                       int Bc, int M, int Ctot, int Cpad, int W,
                       _Float16* __restrict__ X)
{
    long long i = (long long)blockIdx.x * blockDim.x + threadIdx.x;
    long long tot = (long long)Bc * M * Cpad;
    if (i >= tot) return;
    int c = (int)(i % Cpad);
    long long r = i / Cpad;
    int m = (int)(r % M);
    int b = (int)(r / M);
    float v = 0.f;
    if (c < Ctot) {
        if (c < s0.C) v = pk_fetch(s0, b, m, c, M, W);
        else {
            c -= s0.C;
            if (c < s1.C) v = pk_fetch(s1, b, m, c, M, W);
            else {
                c -= s1.C;
                if (c < s2.C) v = pk_fetch(s2, b, m, c, M, W);
                else { c -= s2.C; v = pk_fetch(s3, b, m, c, M, W); }
            }
        }
    }
    X[i] = (_Float16)v;
}

// ---------------------------------------------------------------------------
// KNN (k=3, squared distance) : A [b][3][N] vs nodes Bn [b][3][Mn] -> idx [b][N][3]
// ---------------------------------------------------------------------------
__global__ void k_knn(const float* __restrict__ A, const float* __restrict__ Bn,
                      int N, int Mn, int* __restrict__ idx)
{
    int i = blockIdx.x * blockDim.x + threadIdx.x;
    int b = blockIdx.y;
    if (i >= N) return;
    float ax = A[((long long)b * 3 + 0) * N + i];
    float ay = A[((long long)b * 3 + 1) * N + i];
    float az = A[((long long)b * 3 + 2) * N + i];
    float d0 = 1e30f, d1 = 1e30f, d2 = 1e30f;
    int i0 = 0, i1 = 0, i2 = 0;
    for (int m = 0; m < Mn; ++m) {
        float dx = ax - Bn[((long long)b * 3 + 0) * Mn + m];
        float dy = ay - Bn[((long long)b * 3 + 1) * Mn + m];
        float dz = az - Bn[((long long)b * 3 + 2) * Mn + m];
        float d = dx * dx + dy * dy + dz * dz;
        if (d < d0)      { d2 = d1; i2 = i1; d1 = d0; i1 = i0; d0 = d; i0 = m; }
        else if (d < d1) { d2 = d1; i2 = i1; d1 = d;  i1 = m; }
        else if (d < d2) { d2 = d;  i2 = m; }
    }
    int* o = idx + ((long long)b * N + i) * 3;
    o[0] = i0; o[1] = i1; o[2] = i2;
}

// ---------------------------------------------------------------------------
// Inverse-distance interpolation:
//   w_k = 1 - d_k/sum(d);  Out[b][c][n] = sum_k w_k * F[b][idx_k][c]
// F layout [b][Mn][C] (channel contiguous), Out layout [b][C][N]
// ---------------------------------------------------------------------------
__global__ void k_interp(const float* __restrict__ A, const float* __restrict__ Bn,
                         const int* __restrict__ idx, const float* __restrict__ F,
                         int N, int Mn, int C, float* __restrict__ Out)
{
    int i = blockIdx.x * blockDim.x + threadIdx.x;
    int b = blockIdx.y;
    if (i >= N) return;
    const int* ip = idx + ((long long)b * N + i) * 3;
    float ax = A[((long long)b * 3 + 0) * N + i];
    float ay = A[((long long)b * 3 + 1) * N + i];
    float az = A[((long long)b * 3 + 2) * N + i];
    float dd[3]; int im[3];
    float dsum = 0.f;
    #pragma unroll
    for (int k = 0; k < 3; ++k) {
        int m = ip[k]; im[k] = m;
        float dx = ax - Bn[((long long)b * 3 + 0) * Mn + m];
        float dy = ay - Bn[((long long)b * 3 + 1) * Mn + m];
        float dz = az - Bn[((long long)b * 3 + 2) * Mn + m];
        dd[k] = sqrtf(dx * dx + dy * dy + dz * dz);
        dsum += dd[k];
    }
    float w0 = 1.f - dd[0] / dsum, w1 = 1.f - dd[1] / dsum, w2 = 1.f - dd[2] / dsum;
    const float* f0 = F + ((long long)b * Mn + im[0]) * C;
    const float* f1 = F + ((long long)b * Mn + im[1]) * C;
    const float* f2 = F + ((long long)b * Mn + im[2]) * C;
    float* o = Out + (long long)b * C * N + i;
    for (int c = 0; c < C; c += 4) {
        float4 a4 = *(const float4*)(f0 + c);
        float4 b4 = *(const float4*)(f1 + c);
        float4 c4 = *(const float4*)(f2 + c);
        o[(long long)(c + 0) * N] = w0 * a4.x + w1 * b4.x + w2 * c4.x;
        o[(long long)(c + 1) * N] = w0 * a4.y + w1 * b4.y + w2 * c4.y;
        o[(long long)(c + 2) * N] = w0 * a4.z + w1 * b4.z + w2 * c4.z;
        o[(long long)(c + 3) * N] = w0 * a4.w + w1 * b4.w + w2 * c4.w;
    }
}

// ---------------------------------------------------------------------------
extern "C" void kernel_launch(void* const* d_in, const int* in_sizes, int n_in,
                              void* d_out, int out_size, void* d_ws, size_t ws_size,
                              hipStream_t stream)
{
    (void)in_sizes; (void)n_in; (void)out_size; (void)ws_size;
    const int Bc = 2, N = 20480, MA = 128, MB = 64;
    const int P32 = 80, P16 = 320, P8 = 1280, P4 = 5120;   // H*W per scale

    const float* pc    = (const float*)d_in[0];
    const float* nodeA = (const float*)d_in[1];
    const float* nodeB = (const float*)d_in[2];
    const float* s32   = (const float*)d_in[3];   // [b][512][80]
    const float* s16   = (const float*)d_in[4];   // [b][256][320]
    const float* s8    = (const float*)d_in[5];   // [b][128][1280]
    const float* s4    = (const float*)d_in[6];   // [b][64][5120]
    const float* s2    = (const float*)d_in[7];   // [b][64][20480]
    const float* ig    = (const float*)d_in[8];   // [b][512]
    const float* gf    = (const float*)d_in[9];   // [b][512]
    const float* nbf   = (const float*)d_in[10];  // [b][256][64]
    const float* naf   = (const float*)d_in[11];  // [b][64][128]
    const int*   paIdx = (const int*)d_in[12];    // [b][N][3]
    // params in _PARAM_SPECS insertion order
    const float* Wnba0 = (const float*)d_in[13];  // (256,768)
    const float* Wnba1 = (const float*)d_in[14];  // (80,256)
    const float* Wnbp0 = (const float*)d_in[15];  // (1024,1792)
    const float* Wnbp1 = (const float*)d_in[16];  // (512,1024)
    const float* Wnbp2 = (const float*)d_in[17];  // (512,512)
    const float* Wnaa0 = (const float*)d_in[18];  // (256,576)
    const float* Wnaa1 = (const float*)d_in[19];  // (320,256)
    const float* Wnap0 = (const float*)d_in[20];  // (512,832)
    const float* Wnap1 = (const float*)d_in[21];  // (128,512)
    const float* Wnap2 = (const float*)d_in[22];  // (128,128)
    const float* Wi320 = (const float*)d_in[23];  // (512,1024)
    const float* Wi321 = (const float*)d_in[24];  // (512,512)
    const float* Wi322 = (const float*)d_in[25];  // (64,512)
    const float* Wi160 = (const float*)d_in[26];  // (256,768)
    const float* Wi161 = (const float*)d_in[27];  // (256,256)
    const float* Wi162 = (const float*)d_in[28];  // (128,256)
    const float* Wup1  = (const float*)d_in[29];  // (256,1088)
    const float* Wup2  = (const float*)d_in[30];  // (128,384)
    const float* Wup3  = (const float*)d_in[31];  // (64,256)

    float* outImg = (float*)d_out;                       // [b][64][5120]
    float* outPa  = outImg + (long long)Bc * 64 * P4;    // [b][128][N]
    float* outPb  = outPa + (long long)Bc * 128 * N;     // [b][512][N]

    // ---- workspace bump allocator ----
    size_t off = 0;
    auto alloc = [&](size_t bytes) -> void* {
        size_t cur = (off + 255) & ~(size_t)255;
        off = cur + bytes;
        return (void*)((char*)d_ws + cur);
    };
    auto aH = [&](size_t n) -> _Float16* { return (_Float16*)alloc(n * 2); };
    auto aF = [&](size_t n) -> float*    { return (float*)alloc(n * 4); };

    float* stats = aF(2 * 1024);
    float* Yt    = aF((size_t)Bc * 5120 * 64);   // largest transient GEMM output
    float* E32   = aF((size_t)Bc * 80 * 256);
    float* E16   = aF((size_t)Bc * 320 * 64);
    float* NBW   = aF((size_t)Bc * 64 * 512);
    float* NAW   = aF((size_t)Bc * 128 * 256);
    float* UPNB  = aF((size_t)Bc * 64 * 512);
    float* UPNA  = aF((size_t)Bc * 128 * 128);
    float* ABf   = aF((size_t)Bc * 512 * 128);
    int*   idxPB = (int*)alloc((size_t)Bc * N * 3 * 4);
    int*   idxAB = (int*)alloc((size_t)Bc * 128 * 3 * 4);

    _Float16* Xi32a = aH((size_t)Bc * 80 * 1024);
    _Float16* Xi32b = aH((size_t)Bc * 80 * 512);
    _Float16* A32   = aH((size_t)Bc * 80 * 64);
    _Float16* Fus32 = aH((size_t)Bc * 80 * 768);
    _Float16* Xi16a = aH((size_t)Bc * 320 * 768);
    _Float16* Xi16b = aH((size_t)Bc * 320 * 256);
    _Float16* A16   = aH((size_t)Bc * 320 * 128);
    _Float16* Xu1   = aH((size_t)Bc * 320 * 1088);
    _Float16* U1    = aH((size_t)Bc * 320 * 256);
    _Float16* Xu2   = aH((size_t)Bc * 1280 * 384);
    _Float16* U2    = aH((size_t)Bc * 1280 * 128);
    _Float16* Xu3   = aH((size_t)Bc * 5120 * 256);
    _Float16* Xnba  = aH((size_t)Bc * 64 * 768);
    _Float16* Xnbb  = aH((size_t)Bc * 64 * 256);
    _Float16* Anb   = aH((size_t)Bc * 64 * 96);
    _Float16* Xnp0  = aH((size_t)Bc * 64 * 1792);
    _Float16* Xnp1  = aH((size_t)Bc * 64 * 1024);
    _Float16* Xnp2  = aH((size_t)Bc * 64 * 512);
    _Float16* Xnaa  = aH((size_t)Bc * 128 * 576);
    _Float16* Xnab  = aH((size_t)Bc * 128 * 256);
    _Float16* Ana   = aH((size_t)Bc * 128 * 320);
    _Float16* Xap0  = aH((size_t)Bc * 128 * 832);
    _Float16* Xap1  = aH((size_t)Bc * 128 * 512);
    _Float16* Xap2  = aH((size_t)Bc * 128 * 128);

    // ---- launch helpers ----
    auto gemm = [&](const float* W, long long wSb, int Kw,
                    const _Float16* X, int Kpad, float* Y, int O, int M,
                    bool guard) {
        dim3 g((M + 31) / 32, (O + 63) / 64, Bc);
        if (guard)
            k_gemm<true><<<g, dim3(128), 0, stream>>>(W, wSb, Kw, X, Kpad, Y, O, M);
        else
            k_gemm<false><<<g, dim3(128), 0, stream>>>(W, wSb, Kw, X, Kpad, Y, O, M);
    };
    auto bnrelu = [&](const float* Y, int O, int M, _Float16* Xn) {
        k_bn_stats<<<dim3((O + 255) / 256), dim3(256), 0, stream>>>(Y, Bc * M, O, stats);
        long long tot = (long long)Bc * M * O;
        k_bn_relu_pack<<<dim3((unsigned)((tot + 255) / 256)), dim3(256), 0, stream>>>(
            Y, stats, Bc * M, O, Xn, O);
    };
    auto softmaxp = [&](const float* Y, int M, int C, _Float16* Xn, int Cpad) {
        k_softmax_pack<<<dim3(Bc * M), dim3(32), 0, stream>>>(Y, C, Xn, Cpad);
    };
    auto S = [](const void* p, int C, int mode, int p0) {
        Src s; s.p = p; s.C = C; s.mode = mode; s.p0 = p0; return s;
    };
    Src ZS = S(nullptr, 0, 0, 0);
    auto pack = [&](Src a, Src b, Src c, Src d, int M, int Ctot, int Cpad, int W,
                    _Float16* X) {
        long long tot = (long long)Bc * M * Cpad;
        k_pack<<<dim3((unsigned)((tot + 255) / 256)), dim3(256), 0, stream>>>(
            a, b, c, d, Bc, M, Ctot, Cpad, W, X);
    };

    // ================= image branch: 32x attention =================
    pack(S(s32, 512, 0, 0), S(gf, 512, 2, 0), ZS, ZS, P32, 1024, 1024, 1, Xi32a);
    gemm(Wi320, 0, 1024, Xi32a, 1024, Yt, 512, P32, false);  bnrelu(Yt, 512, P32, Xi32b);
    gemm(Wi321, 0, 512,  Xi32b, 512,  Yt, 512, P32, false);  bnrelu(Yt, 512, P32, Xi32b);
    gemm(Wi322, 0, 512,  Xi32b, 512,  Yt, 64,  P32, false);
    softmaxp(Yt, P32, 64, A32, 64);
    // einsum 'bmhw,bcm' : W = node_b_features (per-batch)
    gemm(nbf, (long long)256 * MB, MB, A32, 64, E32, 256, P32, false);
    pack(S(E32, 256, 1, 0), S(s32, 512, 0, 0), ZS, ZS, P32, 768, 768, 1, Fus32);

    // ================= image branch: 16x attention =================
    pack(S(s16, 256, 0, 0), S(gf, 512, 2, 0), ZS, ZS, P16, 768, 768, 1, Xi16a);
    gemm(Wi160, 0, 768, Xi16a, 768, Yt, 256, P16, false);  bnrelu(Yt, 256, P16, Xi16b);
    gemm(Wi161, 0, 256, Xi16b, 256, Yt, 256, P16, false);  bnrelu(Yt, 256, P16, Xi16b);
    gemm(Wi162, 0, 256, Xi16b, 256, Yt, 128, P16, false);
    softmaxp(Yt, P16, 128, A16, 128);
    gemm(naf, (long long)64 * MA, MA, A16, 128, E16, 64, P16, false);

    // ================= up blocks =================
    pack(S(Fus32, 768, 3, 768), S(E16, 64, 1, 0), S(s16, 256, 0, 0), ZS,
         P16, 1088, 1088, 32, Xu1);
    gemm(Wup1, 0, 1088, Xu1, 1088, Yt, 256, P16, false);  bnrelu(Yt, 256, P16, U1);

    pack(S(U1, 256, 3, 256), S(s8, 128, 0, 0), ZS, ZS, P8, 384, 384, 64, Xu2);
    gemm(Wup2, 0, 384, Xu2, 384, Yt, 128, P8, false);  bnrelu(Yt, 128, P8, U2);

    pack(S(U2, 128, 3, 128), S(s4, 64, 0, 0), S(s2, 64, 4, 0), ZS,
         P4, 256, 256, 128, Xu3);
    gemm(Wup3, 0, 256, Xu3, 256, Yt, 64, P4, false);
    k_bn_stats<<<dim3(1), dim3(256), 0, stream>>>(Yt, Bc * P4, 64, stats);
    {
        long long tot = (long long)Bc * 64 * P4;
        k_bn_relu_tr<<<dim3((unsigned)((tot + 255) / 256)), dim3(256), 0, stream>>>(
            Yt, stats, P4, 64, outImg, Bc);
    }

    // ================= point branch: node_b =================
    pack(S(nbf, 256, 0, 0), S(ig, 512, 2, 0), ZS, ZS, MB, 768, 768, 1, Xnba);
    gemm(Wnba0, 0, 768, Xnba, 768, Yt, 256, MB, false);  bnrelu(Yt, 256, MB, Xnbb);
    gemm(Wnba1, 0, 256, Xnbb, 256, Yt, 80, MB, false);
    softmaxp(Yt, MB, 80, Anb, 96);
    // nb_weighted = s32_flat x nb_attn  (W = s32 per-batch, Kw=80, Kpad=96 -> GUARD)
    gemm(s32, (long long)512 * 80, 80, Anb, 96, NBW, 512, MB, true);

    pack(S(nbf, 256, 0, 0), S(gf, 512, 2, 0), S(NBW, 512, 1, 0), S(ig, 512, 2, 0),
         MB, 1792, 1792, 1, Xnp0);
    gemm(Wnbp0, 0, 1792, Xnp0, 1792, Yt, 1024, MB, false);  bnrelu(Yt, 1024, MB, Xnp1);
    gemm(Wnbp1, 0, 1024, Xnp1, 1024, Yt, 512,  MB, false);  bnrelu(Yt, 512,  MB, Xnp2);
    gemm(Wnbp2, 0, 512,  Xnp2, 512,  UPNB, 512, MB, false);

    // pb = interp(knn(pc, node_b), pc, node_b, up_nb)
    k_knn<<<dim3((N + 255) / 256, Bc), dim3(256), 0, stream>>>(pc, nodeB, N, MB, idxPB);
    k_interp<<<dim3((N + 255) / 256, Bc), dim3(256), 0, stream>>>(
        pc, nodeB, idxPB, UPNB, N, MB, 512, outPb);

    // ================= point branch: node_a =================
    pack(S(naf, 64, 0, 0), S(ig, 512, 2, 0), ZS, ZS, MA, 576, 576, 1, Xnaa);
    gemm(Wnaa0, 0, 576, Xnaa, 576, Yt, 256, MA, false);  bnrelu(Yt, 256, MA, Xnab);
    gemm(Wnaa1, 0, 256, Xnab, 256, Yt, 320, MA, false);
    softmaxp(Yt, MA, 320, Ana, 320);
    gemm(s16, (long long)256 * 320, 320, Ana, 320, NAW, 256, MA, false);

    // ab = interp(knn(node_a, node_b), node_a, node_b, up_nb) -> [b][512][128]
    k_knn<<<dim3(1, Bc), dim3(256), 0, stream>>>(nodeA, nodeB, MA, MB, idxAB);
    k_interp<<<dim3(1, Bc), dim3(256), 0, stream>>>(
        nodeA, nodeB, idxAB, UPNB, MA, MB, 512, ABf);

    pack(S(naf, 64, 0, 0), S(ABf, 512, 0, 0), S(NAW, 256, 1, 0), ZS,
         MA, 832, 832, 1, Xap0);
    gemm(Wnap0, 0, 832, Xap0, 832, Yt, 512, MA, false);  bnrelu(Yt, 512, MA, Xap1);
    gemm(Wnap1, 0, 512, Xap1, 512, Yt, 128, MA, false);  bnrelu(Yt, 128, MA, Xap2);
    gemm(Wnap2, 0, 128, Xap2, 128, UPNA, 128, MA, false);

    // pa = interp(node_a_min_k_idx, pc, node_a, up_na)
    k_interp<<<dim3((N + 255) / 256, Bc), dim3(256), 0, stream>>>(
        pc, nodeA, paIdx, UPNA, N, MA, 128, outPa);
}
